// ContrastiveLoss_76072460747474
// MI455X (gfx1250) — matrix operand order repaired and checked by reading, not compiled, not run on hardware
//
#include <hip/hip_runtime.h>
#include <math.h>

// Problem constants
#define NB   128
#define NR   36
#define NW   32
#define ND   256
#define RPAD 48    // M padding for GEMM1 (3 tiles of 16)
#define KPAD 64    // K padding for GEMM2 (2 bf16 k-steps of 32)

#define MARGIN 0.05f
#define TEMP   14.0f
#define LAM    9.0f
#define EPSV   1e-8f

typedef __bf16 bf16_t;
typedef bf16_t v16bf __attribute__((ext_vector_type(16)));
typedef float  v8f   __attribute__((ext_vector_type(8)));
typedef unsigned int u32x4 __attribute__((ext_vector_type(4)));

union FragU { u32x4 q[2]; v16bf v; };

__device__ __forceinline__ unsigned short f2bf(float f) {
  unsigned int u = __float_as_uint(f);
  u += 0x7FFFu + ((u >> 16) & 1u);   // round-to-nearest-even
  return (unsigned short)(u >> 16);
}
__device__ __forceinline__ float bf2f(unsigned short h) {
  return __uint_as_float(((unsigned int)h) << 16);
}
__device__ __forceinline__ v8f vzero() {
  v8f z;
#pragma unroll
  for (int i = 0; i < 8; ++i) z[i] = 0.0f;
  return z;
}
__device__ __forceinline__ float lrelu(float x) { return x >= 0.0f ? x : 0.1f * x; }

// ---------------- prep kernels: bf16 conversion / transpose / norms ----------------

// im_bf[b][r(48)][d(256)], rows >= 36 zero
__global__ void prep_im(const float* __restrict__ im, unsigned short* __restrict__ im_bf) {
  int idx = blockIdx.x * 256 + threadIdx.x;          // NB*RPAD*ND
  int d = idx & (ND - 1);
  int r = (idx >> 8) % RPAD;
  int b = idx / (RPAD * ND);
  float v = (r < NR) ? im[(b * NR + r) * ND + d] : 0.0f;
  im_bf[idx] = f2bf(v);
}

// imT_bf[b][d(256)][r(64)], r >= 36 zero
__global__ void prep_imT(const float* __restrict__ im, unsigned short* __restrict__ imT_bf) {
  int idx = blockIdx.x * 256 + threadIdx.x;          // NB*ND*KPAD
  int r = idx & (KPAD - 1);
  int d = (idx >> 6) & (ND - 1);
  int b = idx >> 14;
  float v = (r < NR) ? im[(b * NR + r) * ND + d] : 0.0f;
  imT_bf[idx] = f2bf(v);
}

// cap_bf[c][w(32)][d(256)] = s * (w < s_l[c])
__global__ void prep_cap(const float* __restrict__ s, const int* __restrict__ s_l,
                         unsigned short* __restrict__ cap_bf) {
  int idx = blockIdx.x * 256 + threadIdx.x;          // NB*NW*ND
  int w = (idx >> 8) & (NW - 1);
  int c = idx >> 13;
  float v = (w < s_l[c]) ? s[idx] : 0.0f;
  cap_bf[idx] = f2bf(v);
}

// n1[c][w] = ||masked caption word|| (f32)
__global__ void prep_n1(const float* __restrict__ s, const int* __restrict__ s_l,
                        float* __restrict__ n1) {
  int idx = blockIdx.x * 256 + threadIdx.x;          // NB*NW
  int w = idx & (NW - 1);
  int c = idx >> 5;
  float acc = 0.0f;
  if (w < s_l[c]) {
    const float* row = s + (c * NW + w) * ND;
    for (int d = 0; d < ND; ++d) { float v = row[d]; acc += v * v; }
  }
  n1[idx] = sqrtf(acc);
}

// ---------------- main kernel: one wave per (image b, caption c) pair ----------------

__device__ __forceinline__ v8f wmma_bf16(const v16bf& a, const v16bf& b, const v8f& c) {
  return __builtin_amdgcn_wmma_f32_16x16x32_bf16(false, a, false, b, (short)0, c, false, false);
}

__global__ __launch_bounds__(256) void xattn_scores_kernel(
    const unsigned short* __restrict__ im_bf,
    const unsigned short* __restrict__ imT_bf,
    const unsigned short* __restrict__ cap_bf,
    const float* __restrict__ n1,
    const float* __restrict__ im_mask,
    const int* __restrict__ s_l,
    float* __restrict__ scores) {
  __shared__ __align__(16) unsigned short y_lds[8][NW][KPAD];   // 32 KB

  const int lane = threadIdx.x & 31;
  const int wv   = threadIdx.x >> 5;
  const int pid  = blockIdx.x * 8 + wv;
  const int b    = pid >> 7;       // image index (shared by all 8 waves of a block)
  const int c    = pid & 127;      // caption index
  const int half = lane >> 4;      // 0: lanes 0-15, 1: lanes 16-31
  const int lm   = lane & 15;

  const unsigned short* imb  = im_bf  + b * (RPAD * ND);
  const unsigned short* capc = cap_bf + c * (NW * ND);
  const unsigned short* imTb = imT_bf + b * (ND * KPAD);

  // ---------------- GEMM1: attn = im[b] (48x256) x cap[c]^T (256x32), bf16 WMMA ----
  v8f acc[3][2];
#pragma unroll
  for (int mi = 0; mi < 3; ++mi)
#pragma unroll
    for (int ni = 0; ni < 2; ++ni) acc[mi][ni] = vzero();

#pragma unroll
  for (int k = 0; k < 8; ++k) {             // K = 256 = 8 * 32
    FragU a[3], bb[2];
#pragma unroll
    for (int mi = 0; mi < 3; ++mi) {        // A: row = mi*16 + lm, 16 contiguous K
      const u32x4* p = (const u32x4*)(imb + (mi * 16 + lm) * ND + k * 32 + half * 16);
      a[mi].q[0] = p[0];
      a[mi].q[1] = p[1];                    // +8 bf16 elements
    }
#pragma unroll
    for (int ni = 0; ni < 2; ++ni) {        // B: col w = ni*16 + lm, 16 contiguous K
      const u32x4* p = (const u32x4*)(capc + (ni * 16 + lm) * ND + k * 32 + half * 16);
      bb[ni].q[0] = p[0];
      bb[ni].q[1] = p[1];
    }
#pragma unroll
    for (int mi = 0; mi < 3; ++mi)
#pragma unroll
      for (int ni = 0; ni < 2; ++ni)
        acc[mi][ni] = wmma_bf16(a[mi].v, bb[ni].v, acc[mi][ni]);
  }

  // ----- leaky relu + masked l2-normalize over W per region row, fold *LAM in -----
  // element (mi, ni, j): r = mi*16 + half*8 + j, w = ni*16 + lm
  float x[3][2][8];
  float mval[3][8];
#pragma unroll
  for (int mi = 0; mi < 3; ++mi) {
#pragma unroll
    for (int j = 0; j < 8; ++j) {
      int r = mi * 16 + half * 8 + j;
      float m = (r < NR) ? im_mask[b * NR + r] : 0.0f;
      mval[mi][j] = m;
      float v0 = lrelu(acc[mi][0][j]) + (1.0f - m);
      float v1 = lrelu(acc[mi][1][j]) + (1.0f - m);
      float ss = v0 * v0 + v1 * v1;         // all W values for row r live in this half
      ss += __shfl_xor(ss, 1, 32);
      ss += __shfl_xor(ss, 2, 32);
      ss += __shfl_xor(ss, 4, 32);
      ss += __shfl_xor(ss, 8, 32);
      float inv = m * LAM / (sqrtf(ss) + EPSV);   // attn = X/nrm*m, x = attn*LAM
      x[mi][0][j] = v0 * inv;
      x[mi][1][j] = v1 * inv;
    }
  }

  // ----- softmax over regions r per word w (cross-half reduction, xor 16) -----
#pragma unroll
  for (int ni = 0; ni < 2; ++ni) {
    float mx = -1e30f;
#pragma unroll
    for (int mi = 0; mi < 3; ++mi)
#pragma unroll
      for (int j = 0; j < 8; ++j) {
        int r = mi * 16 + half * 8 + j;
        if (r < NR) mx = fmaxf(mx, x[mi][ni][j]);
      }
    mx = fmaxf(mx, __shfl_xor(mx, 16, 32));
    float se = 0.0f;
#pragma unroll
    for (int mi = 0; mi < 3; ++mi)
#pragma unroll
      for (int j = 0; j < 8; ++j) {
        int r = mi * 16 + half * 8 + j;
        float e = (r < NR) ? __expf(x[mi][ni][j] - mx) * mval[mi][j] : 0.0f;
        x[mi][ni][j] = e;
        se += e;
      }
    se += __shfl_xor(se, 16, 32);
    float rinv = 1.0f / fmaxf(se, 1e-30f);
#pragma unroll
    for (int mi = 0; mi < 3; ++mi)
#pragma unroll
      for (int j = 0; j < 8; ++j) x[mi][ni][j] *= rinv;
  }

  // ----- stage y (32x64 bf16, K-padded) into LDS in A-fragment-friendly layout -----
  {
    u32x4 z; z.x = 0; z.y = 0; z.z = 0; z.w = 0;
    u32x4* zp = (u32x4*)&y_lds[wv][lane][48];   // zero K pad 48..63 (w = lane)
    zp[0] = z; zp[1] = z;
  }
#pragma unroll
  for (int mi = 0; mi < 3; ++mi)
#pragma unroll
    for (int ni = 0; ni < 2; ++ni)
#pragma unroll
      for (int j = 0; j < 8; ++j) {
        int r = mi * 16 + half * 8 + j;
        int w = ni * 16 + lm;
        y_lds[wv][w][r] = f2bf(x[mi][ni][j]);
      }
  __syncthreads();

  // A fragments of y: M = w, K = r (two 16-wide M tiles, two 32-wide K steps)
  FragU ya[2][2];
#pragma unroll
  for (int wi = 0; wi < 2; ++wi)
#pragma unroll
    for (int ki = 0; ki < 2; ++ki) {
      const u32x4* p = (const u32x4*)&y_lds[wv][wi * 16 + lm][ki * 32 + half * 16];
      ya[wi][ki].q[0] = p[0];
      ya[wi][ki].q[1] = p[1];
    }

  // ---------------- GEMM2: wctx = y (32x64) x imT (64x256), streamed over D-tiles ---
  float w12s[2][8], n2s[2][8];
#pragma unroll
  for (int wi = 0; wi < 2; ++wi)
#pragma unroll
    for (int j = 0; j < 8; ++j) { w12s[wi][j] = 0.0f; n2s[wi][j] = 0.0f; }

#pragma unroll
  for (int ni = 0; ni < 16; ++ni) {          // 16 tiles of 16 columns of D
    v8f acc2[2] = { vzero(), vzero() };
#pragma unroll
    for (int ki = 0; ki < 2; ++ki) {
      FragU bb;                              // B: col d = ni*16 + lm, 16 contiguous r
      const u32x4* p = (const u32x4*)(imTb + (ni * 16 + lm) * KPAD + ki * 32 + half * 16);
      bb.q[0] = p[0];
      bb.q[1] = p[1];
      acc2[0] = wmma_bf16(ya[0][ki].v, bb.v, acc2[0]);
      acc2[1] = wmma_bf16(ya[1][ki].v, bb.v, acc2[1]);
    }
    int d = ni * 16 + lm;                    // element: w = wi*16 + half*8 + j, col d
#pragma unroll
    for (int wi = 0; wi < 2; ++wi)
#pragma unroll
      for (int j = 0; j < 8; ++j) {
        int w = wi * 16 + half * 8 + j;
        float v  = acc2[wi][j];
        float cv = bf2f(capc[w * ND + d]);
        w12s[wi][j] += cv * v;
        n2s[wi][j]  += v * v;
      }
  }

  // ----- cosine + masked mean over words -> scores[b][c] -----
  const int sl = s_l[c];
  float partial = 0.0f;
#pragma unroll
  for (int wi = 0; wi < 2; ++wi)
#pragma unroll
    for (int j = 0; j < 8; ++j) {
      float a = w12s[wi][j];
      float q = n2s[wi][j];
      a += __shfl_xor(a, 1, 32); a += __shfl_xor(a, 2, 32);
      a += __shfl_xor(a, 4, 32); a += __shfl_xor(a, 8, 32);
      q += __shfl_xor(q, 1, 32); q += __shfl_xor(q, 2, 32);
      q += __shfl_xor(q, 4, 32); q += __shfl_xor(q, 8, 32);
      int w = wi * 16 + half * 8 + j;
      float denom = fmaxf(n1[c * NW + w] * sqrtf(q), EPSV);
      float cosw = a / denom;
      if (w < sl) partial += cosw;
    }
  partial *= (1.0f / 16.0f);                 // 16 duplicate lanes per half
  partial += __shfl_xor(partial, 1, 32);
  partial += __shfl_xor(partial, 2, 32);
  partial += __shfl_xor(partial, 4, 32);
  partial += __shfl_xor(partial, 8, 32);
  partial += __shfl_xor(partial, 16, 32);
  if (lane == 0) scores[b * NB + c] = partial / (float)sl;
}

// ---------------- contrastive loss over the 128x128 score matrix ----------------

__global__ __launch_bounds__(128) void loss_kernel(const float* __restrict__ scores,
                                                   const int* __restrict__ qid,
                                                   float* __restrict__ out) {
  __shared__ float sval[NB];
  __shared__ float scnt[NB];
  int i = threadIdx.x;
  int q = qid[i];
  bool dup = false;
  for (int j = 0; j < i; ++j)
    if (qid[j] == q) dup = true;

  const float* row = scores + i * NB;
  float mx = -1e30f;
  for (int j = 0; j < NB; ++j) {
    float l = (row[j] - ((i == j) ? MARGIN : 0.0f)) * TEMP;
    mx = fmaxf(mx, l);
  }
  float se = 0.0f;
  for (int j = 0; j < NB; ++j) {
    float l = (row[j] - ((i == j) ? MARGIN : 0.0f)) * TEMP;
    se += __expf(l - mx);
  }
  float logZ = __logf(se) + mx;
  float picked = (row[i] - MARGIN) * TEMP;
  float valid = dup ? 0.0f : 1.0f;
  sval[i] = (logZ - picked) * valid;
  scnt[i] = valid;
  __syncthreads();
  if (i == 0) {
    float a = 0.0f, n = 0.0f;
    for (int k = 0; k < NB; ++k) { a += sval[k]; n += scnt[k]; }
    out[0] = a / fmaxf(n, 1.0f);
  }
}

// ---------------- launch ----------------

extern "C" void kernel_launch(void* const* d_in, const int* in_sizes, int n_in,
                              void* d_out, int out_size, void* d_ws, size_t ws_size,
                              hipStream_t stream) {
  (void)in_sizes; (void)n_in; (void)out_size; (void)ws_size;
  const float* im      = (const float*)d_in[0];   // (128,36,256)
  const float* s       = (const float*)d_in[1];   // (128,32,256)
  const int*   s_l     = (const int*)d_in[2];     // (128,)
  const float* im_mask = (const float*)d_in[3];   // (128,36)
  const int*   qid     = (const int*)d_in[4];     // (128,)
  // d_in[5], d_in[6] (class_scores/labels) unused by the reference output.

  char* ws = (char*)d_ws;
  size_t off = 0;
  unsigned short* im_bf  = (unsigned short*)(ws + off); off += (size_t)NB * RPAD * ND * 2; // 3.0 MB
  unsigned short* imT_bf = (unsigned short*)(ws + off); off += (size_t)NB * ND * KPAD * 2; // 4.0 MB
  unsigned short* cap_bf = (unsigned short*)(ws + off); off += (size_t)NB * NW * ND * 2;   // 2.0 MB
  float* n1     = (float*)(ws + off); off += (size_t)NB * NW * 4;
  float* scores = (float*)(ws + off); off += (size_t)NB * NB * 4;   // total ~9.4 MB

  prep_im <<<(NB * RPAD * ND) / 256, 256, 0, stream>>>(im, im_bf);
  prep_imT<<<(NB * ND * KPAD) / 256, 256, 0, stream>>>(im, imT_bf);
  prep_cap<<<(NB * NW * ND) / 256, 256, 0, stream>>>(s, s_l, cap_bf);
  prep_n1 <<<(NB * NW) / 256, 256, 0, stream>>>(s, s_l, n1);

  // one wave per (b,c) pair: 128*128 pairs / 8 waves per block
  xattn_scores_kernel<<<(NB * NB) / 8, 256, 0, stream>>>(
      im_bf, imT_bf, cap_bf, n1, im_mask, s_l, scores);

  loss_kernel<<<1, NB, 0, stream>>>(scores, qid, (float*)d_out);
}
